// VanillaMultiHeadAttention_31086973288775
// MI455X (gfx1250) — matrix operand order repaired and checked by reading, not compiled, not run on hardware
//
#include <hip/hip_runtime.h>
#include <math.h>

// Problem constants (reference: B=4, L=2048, D=1024, H=16, HD=64)
#define BB 4
#define LL 2048
#define DD 1024
#define HH 16
#define HDIM 64
#define BLROWS (BB * LL)   // 8192

#define LOG2E 1.44269504088896340736f

typedef __bf16 bf16_t;
typedef bf16_t v16bf __attribute__((ext_vector_type(16)));
typedef float  v8f   __attribute__((ext_vector_type(8)));
typedef float  f32x4 __attribute__((ext_vector_type(4)));
typedef unsigned int u32;
typedef u32 u32x4 __attribute__((ext_vector_type(4)));
typedef u32 u32x2 __attribute__((ext_vector_type(2)));

union frag_u { u32x4 q[2]; v16bf v; };

// ---------------------------------------------------------------------------
// WMMA fragment loaders (wave32; A/B k order is this code's contiguous choice, C/D per CDNA5 ISA 7.12.2)
// A (16x32 bf16): lane&15 -> row M, K = 16*(lane>>4) + e. One 32B run -> 2x b128
//   loads.
// B (32x16 bf16): lane&15 -> col N, K = 16*(lane>>4) + e. One 32B run.
// C/D (16x16 f32): slot v -> row M = v + 8*(lane>>4), col N = lane&15.
// ---------------------------------------------------------------------------

__device__ __forceinline__ v16bf load_a_bf16(const bf16_t* p, int ld) {
  const int lane = threadIdx.x & 31;
  const bf16_t* q = p + (size_t)(lane & 15) * ld + ((lane >> 4) << 4);
  frag_u f;
  f.q[0] = *(const u32x4*)q;         // K = kbase .. kbase+7
  f.q[1] = *(const u32x4*)(q + 8);   // K = kbase+8 .. kbase+15
  return f.v;
}

// B = S^T where S is row-major with rows indexed by N (W[n][k], Q[query][hd]).
__device__ __forceinline__ v16bf load_bT_bf16(const bf16_t* p, int ld) {
  const int lane = threadIdx.x & 31;
  const bf16_t* q = p + (size_t)(lane & 15) * ld + ((lane >> 4) << 4);
  frag_u f;
  f.q[0] = *(const u32x4*)q;         // K = kbase .. kbase+7
  f.q[1] = *(const u32x4*)(q + 8);   // K = kbase+8 .. kbase+15
  return f.v;
}

__device__ __forceinline__ v8f wmma_bf16(v16bf a, v16bf b, v8f c) {
  return __builtin_amdgcn_wmma_f32_16x16x32_bf16(false, a, false, b, (short)0, c,
                                                 false, false);
}

__device__ __forceinline__ u32 pk_bf16(float lo, float hi) {
  union { bf16_t b[2]; u32 u; } p;
  p.b[0] = (bf16_t)lo;
  p.b[1] = (bf16_t)hi;
  return p.u;
}

// ---------------------------------------------------------------------------
// Kernel 0: fp32 -> bf16 conversion, vectorized (float4 in, 4x bf16 out)
// ---------------------------------------------------------------------------
__global__ void __launch_bounds__(256)
cvt_f32_bf16_kernel(const float* __restrict__ src, bf16_t* __restrict__ dst,
                    int n4) {
  const int i = blockIdx.x * 256 + threadIdx.x;
  if (i >= n4) return;
  const f32x4 f = ((const f32x4*)src)[i];
  union { bf16_t b[4]; u32x2 u; } o;
  o.b[0] = (bf16_t)f[0];
  o.b[1] = (bf16_t)f[1];
  o.b[2] = (bf16_t)f[2];
  o.b[3] = (bf16_t)f[3];
  ((u32x2*)dst)[i] = o.u;
}

// ---------------------------------------------------------------------------
// Kernel 1: QKV projection (X @ W^T + b), bf16 in, fused RoPE, bf16 out.
// grid (BLROWS/128, H), block 128 (4 waves); wave -> 32 rows x 64 cols (1 head).
// mode 1: RoPE + premul, store (B,H,L,HD).  mode 0: store V^T as (B,H,HD,L).
// ---------------------------------------------------------------------------
__global__ void __launch_bounds__(128)
qkv_proj_rope_kernel(const bf16_t* __restrict__ Xb, const bf16_t* __restrict__ Wb,
                     const float* __restrict__ bias, bf16_t* __restrict__ dst,
                     int mode, float premul) {
  const int lane = threadIdx.x & 31;
  const int wid  = threadIdx.x >> 5;
  const int row0 = (blockIdx.x * 4 + wid) * 32;  // row in flattened (B*L)
  const int h    = blockIdx.y;                   // head == 64-col tile
  const int col0 = h * HDIM;
  const int g    = lane >> 4;
  const int nl   = lane & 15;

  v8f acc[2][4];
#pragma unroll
  for (int mf = 0; mf < 2; ++mf)
#pragma unroll
    for (int t = 0; t < 4; ++t)
#pragma unroll
      for (int i = 0; i < 8; ++i) acc[mf][t][i] = 0.0f;

  for (int k0 = 0; k0 < DD; k0 += 32) {
    const v16bf a0 = load_a_bf16(Xb + (size_t)row0 * DD + k0, DD);
    const v16bf a1 = load_a_bf16(Xb + (size_t)(row0 + 16) * DD + k0, DD);
#pragma unroll
    for (int t = 0; t < 4; ++t) {
      const v16bf b = load_bT_bf16(Wb + (size_t)(col0 + t * 16) * DD + k0, DD);
      acc[0][t] = wmma_bf16(a0, b, acc[0][t]);
      acc[1][t] = wmma_bf16(a1, b, acc[1][t]);
    }
  }

  float bvs[4];
#pragma unroll
  for (int t = 0; t < 4; ++t) bvs[t] = bias[col0 + t * 16 + nl];

#pragma unroll
  for (int mf = 0; mf < 2; ++mf) {
    float vals[4][8];
#pragma unroll
    for (int t = 0; t < 4; ++t)
#pragma unroll
      for (int v = 0; v < 8; ++v) vals[t][v] = acc[mf][t][v] + bvs[t];

    if (mode == 1) {
      // RoPE: hd pairs (t, t+2) share the angle (freq index hd & 31).
      const float inv0 = powf(10000.0f, -((float)nl) / 32.0f);
      const float inv1 = powf(10000.0f, -((float)(nl + 16)) / 32.0f);
#pragma unroll
      for (int v = 0; v < 8; ++v) {
        const int r = row0 + mf * 16 + v + 8 * g;
        const float pos = (float)(r & (LL - 1));
        const float a0 = pos * inv0, a1 = pos * inv1;
        const float c0 = __cosf(a0), s0 = __sinf(a0);
        const float c1 = __cosf(a1), s1 = __sinf(a1);
        const float x0 = vals[0][v], x1 = vals[1][v];
        const float x2 = vals[2][v], x3 = vals[3][v];
        vals[0][v] = x0 * c0 - x2 * s0;
        vals[1][v] = x1 * c1 - x3 * s1;
        vals[2][v] = x2 * c0 + x0 * s0;
        vals[3][v] = x3 * c1 + x1 * s1;
      }
#pragma unroll
      for (int v = 0; v < 8; ++v) {
        const int r   = row0 + mf * 16 + v + 8 * g;
        const int bi  = r >> 11;
        const int pos = r & (LL - 1);
        const size_t base = (((size_t)bi * HH + h) * LL + pos) * HDIM;
#pragma unroll
        for (int t = 0; t < 4; ++t)
          dst[base + t * 16 + nl] = (bf16_t)(vals[t][v] * premul);
      }
    } else {
      // V^T layout: dst[((bi*H + h)*HD + hd)*L + pos]
#pragma unroll
      for (int v = 0; v < 8; ++v) {
        const int r   = row0 + mf * 16 + v + 8 * g;
        const int bi  = r >> 11;
        const int pos = r & (LL - 1);
#pragma unroll
        for (int t = 0; t < 4; ++t) {
          const int hd = t * 16 + nl;
          dst[(((size_t)bi * HH + h) * HDIM + hd) * LL + pos] = (bf16_t)vals[t][v];
        }
      }
    }
  }
}

// ---------------------------------------------------------------------------
// Kernel 2: causal flash attention with ALiBi, fully transposed formulation.
// S^T = K * Q^T  (A = K rows, B = Q rows -> one query per LANE: softmax
// reductions are per-lane register ops + one shfl_xor(16) with the partner
// lane holding the other key half).  O^T += V^T * P^T (A = V^T rows; P^T is
// already the B operand -- key halves swapped via 4 packed shfl_xor(16)).
// One wave = one 16-query tile; key blocks of 32; Q pre-scaled by
// (1/sqrt(HD))*log2(e) so all exponentials are native exp2.
// grid (L/128, B*H), block 256 (8 waves). Output: bf16 (B*L, D).
// ---------------------------------------------------------------------------
__global__ void __launch_bounds__(256)
attn_kernel(const bf16_t* __restrict__ Qw, const bf16_t* __restrict__ Kw,
            const bf16_t* __restrict__ Vt, bf16_t* __restrict__ attn_out) {
  const int lane = threadIdx.x & 31;
  const int wid  = threadIdx.x >> 5;
  const int g    = lane >> 4;
  const int nl   = lane & 15;
  const int bh   = blockIdx.y;
  const int b    = bh >> 4;
  const int h    = bh & 15;
  const int q0   = (blockIdx.x * 8 + wid) * 16;

  const bf16_t* Qb  = Qw + (size_t)bh * LL * HDIM;
  const bf16_t* Kb  = Kw + (size_t)bh * LL * HDIM;
  const bf16_t* Vtb = Vt + (size_t)bh * HDIM * LL;

  // Q as B-operand (lane -> query column), hoisted out of the key loop.
  const v16bf bq0 = load_bT_bf16(Qb + (size_t)q0 * HDIM, HDIM);       // hd 0..31
  const v16bf bq1 = load_bT_bf16(Qb + (size_t)q0 * HDIM + 32, HDIM);  // hd 32..63

  v8f O[4];  // O^T tiles: row hd = t*16 + v + 8g, col query = nl
#pragma unroll
  for (int t = 0; t < 4; ++t)
#pragma unroll
    for (int i = 0; i < 8; ++i) O[t][i] = 0.0f;

  // ALiBi slope in log2 domain (exp2 softmax)
  const float slope = exp2f(-0.5f * (float)(h + 1)) * LOG2E;
  const int   qi     = q0 + nl;             // this lane's query index
  const float slopei = slope * (float)qi;
  float rmax = -1e30f, rsum = 0.0f;

  for (int k0 = 0; k0 < q0 + 16; k0 += 32) {
    // S^T tiles: St0 keys k0..k0+15, St1 keys k0+16..k0+31 (rows), query cols.
    v8f S0, S1;
#pragma unroll
    for (int i = 0; i < 8; ++i) { S0[i] = 0.0f; S1[i] = 0.0f; }
    {
      v16bf ak = load_a_bf16(Kb + (size_t)k0 * HDIM, HDIM);
      S0 = wmma_bf16(ak, bq0, S0);
      ak = load_a_bf16(Kb + (size_t)k0 * HDIM + 32, HDIM);
      S0 = wmma_bf16(ak, bq1, S0);
      ak = load_a_bf16(Kb + (size_t)(k0 + 16) * HDIM, HDIM);
      S1 = wmma_bf16(ak, bq0, S1);
      ak = load_a_bf16(Kb + (size_t)(k0 + 16) * HDIM + 32, HDIM);
      S1 = wmma_bf16(ak, bq1, S1);
    }

    // This lane's keys: j0 = k0 + 8g + v (St0), j1 = j0 + 16 (St1).
    float p0[8], p1[8];
    float lm = -1e30f;
#pragma unroll
    for (int v = 0; v < 8; ++v) {
      const int j0 = k0 + 8 * g + v;
      const int j1 = j0 + 16;
      p0[v] = (j0 <= qi) ? (S0[v] + slope * (float)j0 - slopei) : -1e30f;
      p1[v] = (j1 <= qi) ? (S1[v] + slope * (float)j1 - slopei) : -1e30f;
      lm = fmaxf(lm, fmaxf(p0[v], p1[v]));
    }
    lm = fmaxf(lm, __shfl_xor(lm, 16, 32));   // combine partner key half
    const float nmax = fmaxf(rmax, lm);
    const float corr = exp2f(rmax - nmax);
    rmax = nmax;

    float e0[8], e1[8], bs = 0.0f;
#pragma unroll
    for (int v = 0; v < 8; ++v) {
      e0[v] = exp2f(p0[v] - nmax);
      e1[v] = exp2f(p1[v] - nmax);
      bs += e0[v] + e1[v];
    }
    bs += __shfl_xor(bs, 16, 32);
    rsum = rsum * corr + bs;

#pragma unroll
    for (int t = 0; t < 4; ++t)
#pragma unroll
      for (int i = 0; i < 8; ++i) O[t][i] *= corr;

    // Build P^T B-fragment: lane needs keys 16g..16g+15 for query nl.
    // Locally owned: keys 8g+v (tile0), 16+8g+v (tile1); the missing half
    // lives in the partner lane (lane ^ 16) -> 4 packed shfl_xor.
    u32 t0p[4], t1p[4], sel[4], oth[4];
#pragma unroll
    for (int jj = 0; jj < 4; ++jj) {
      t0p[jj] = pk_bf16(e0[2 * jj], e0[2 * jj + 1]);
      t1p[jj] = pk_bf16(e1[2 * jj], e1[2 * jj + 1]);
      sel[jj] = g ? t0p[jj] : t1p[jj];
    }
#pragma unroll
    for (int jj = 0; jj < 4; ++jj) oth[jj] = __shfl_xor(sel[jj], 16, 32);

    union { u32 u[8]; v16bf v; } bp;
#pragma unroll
    for (int jj = 0; jj < 4; ++jj) {
      bp.u[jj]     = g ? oth[jj] : t0p[jj];  // B elems 0..7  : keys 16g..16g+7
      bp.u[4 + jj] = g ? t1p[jj] : oth[jj];  // B elems 8..15 : keys 16g+8..+15
    }

    // O^T += V^T * P^T
#pragma unroll
    for (int t = 0; t < 4; ++t) {
      const v16bf av = load_a_bf16(Vtb + (size_t)(t * 16) * LL + k0, LL);
      O[t] = wmma_bf16(av, bp.v, O[t]);
    }
  }

  // Normalize and store: per (lane, t) 8 consecutive hd -> one 16B store.
  const float inv = 1.0f / rsum;
  const size_t r = (size_t)b * LL + q0 + nl;
#pragma unroll
  for (int t = 0; t < 4; ++t) {
    union { u32 u[4]; u32x4 q; } o;
#pragma unroll
    for (int jj = 0; jj < 4; ++jj)
      o.u[jj] = pk_bf16(O[t][2 * jj] * inv, O[t][2 * jj + 1] * inv);
    *(u32x4*)(attn_out + r * DD + h * HDIM + t * 16 + 8 * g) = o.q;
  }
}

// ---------------------------------------------------------------------------
// Kernel 3: output projection attn(bf16) @ Wo^T + bo -> fp32 d_out
// grid (BLROWS/128, D/64), block 128; wave -> 32 rows x 64 cols.
// ---------------------------------------------------------------------------
__global__ void __launch_bounds__(128)
out_proj_kernel(const bf16_t* __restrict__ Xa, const bf16_t* __restrict__ Wb,
                const float* __restrict__ bias, float* __restrict__ out) {
  const int lane = threadIdx.x & 31;
  const int wid  = threadIdx.x >> 5;
  const int row0 = (blockIdx.x * 4 + wid) * 32;
  const int col0 = blockIdx.y * 64;
  const int g    = lane >> 4;
  const int nl   = lane & 15;

  v8f acc[2][4];
#pragma unroll
  for (int mf = 0; mf < 2; ++mf)
#pragma unroll
    for (int t = 0; t < 4; ++t)
#pragma unroll
      for (int i = 0; i < 8; ++i) acc[mf][t][i] = 0.0f;

  for (int k0 = 0; k0 < DD; k0 += 32) {
    const v16bf a0 = load_a_bf16(Xa + (size_t)row0 * DD + k0, DD);
    const v16bf a1 = load_a_bf16(Xa + (size_t)(row0 + 16) * DD + k0, DD);
#pragma unroll
    for (int t = 0; t < 4; ++t) {
      const v16bf b = load_bT_bf16(Wb + (size_t)(col0 + t * 16) * DD + k0, DD);
      acc[0][t] = wmma_bf16(a0, b, acc[0][t]);
      acc[1][t] = wmma_bf16(a1, b, acc[1][t]);
    }
  }

#pragma unroll
  for (int mf = 0; mf < 2; ++mf)
#pragma unroll
    for (int v = 0; v < 8; ++v) {
      const size_t r = (size_t)row0 + mf * 16 + v + 8 * g;
#pragma unroll
      for (int t = 0; t < 4; ++t) {
        const int n = col0 + t * 16 + nl;
        out[r * DD + n] = acc[mf][t][v] + bias[n];
      }
    }
}

// ---------------------------------------------------------------------------
// Launch
// ---------------------------------------------------------------------------
extern "C" void kernel_launch(void* const* d_in, const int* in_sizes, int n_in,
                              void* d_out, int out_size, void* d_ws, size_t ws_size,
                              hipStream_t stream) {
  (void)in_sizes; (void)n_in; (void)out_size; (void)ws_size;

  const float* query = (const float*)d_in[0];
  const float* keyi  = (const float*)d_in[1];
  const float* vali  = (const float*)d_in[2];
  // d_in[3] = causal mask (hard-coded causal in attn_kernel)
  const float* Wq = (const float*)d_in[4];
  const float* bq = (const float*)d_in[5];
  const float* Wk = (const float*)d_in[6];
  const float* bk = (const float*)d_in[7];
  const float* Wv = (const float*)d_in[8];
  const float* bv = (const float*)d_in[9];
  const float* Wo = (const float*)d_in[10];
  const float* bo = (const float*)d_in[11];

  // Workspace layout (all bf16): Xq/Xk/Xv 16MB each, Wq/Wk/Wv/Wo 2MB each,
  // Q/K 16MB each, V^T 16MB. Attention output (bf16, 16MB) aliases Xq, which
  // is no longer read once the Q projection has completed (in-stream order).
  char* ws = (char*)d_ws;
  const size_t xbytes = (size_t)BLROWS * DD * sizeof(bf16_t);   // 16 MB
  const size_t wbytes = (size_t)DD * DD * sizeof(bf16_t);       //  2 MB
  bf16_t* Xqb = (bf16_t*)(ws);
  bf16_t* Xkb = (bf16_t*)(ws + xbytes);
  bf16_t* Xvb = (bf16_t*)(ws + 2 * xbytes);
  bf16_t* Wqb = (bf16_t*)(ws + 3 * xbytes);
  bf16_t* Wkb = (bf16_t*)(ws + 3 * xbytes + wbytes);
  bf16_t* Wvb = (bf16_t*)(ws + 3 * xbytes + 2 * wbytes);
  bf16_t* Wob = (bf16_t*)(ws + 3 * xbytes + 3 * wbytes);
  bf16_t* Qw  = (bf16_t*)(ws + 3 * xbytes + 4 * wbytes);
  bf16_t* Kw  = (bf16_t*)(ws + 4 * xbytes + 4 * wbytes);
  bf16_t* Vt  = (bf16_t*)(ws + 5 * xbytes + 4 * wbytes);
  bf16_t* attnb = Xqb;  // alias: Xq consumed before attn_kernel runs

  const int xn4 = BLROWS * DD / 4;
  const int wn4 = DD * DD / 4;
  cvt_f32_bf16_kernel<<<(xn4 + 255) / 256, 256, 0, stream>>>(query, Xqb, xn4);
  cvt_f32_bf16_kernel<<<(xn4 + 255) / 256, 256, 0, stream>>>(keyi,  Xkb, xn4);
  cvt_f32_bf16_kernel<<<(xn4 + 255) / 256, 256, 0, stream>>>(vali,  Xvb, xn4);
  cvt_f32_bf16_kernel<<<(wn4 + 255) / 256, 256, 0, stream>>>(Wq, Wqb, wn4);
  cvt_f32_bf16_kernel<<<(wn4 + 255) / 256, 256, 0, stream>>>(Wk, Wkb, wn4);
  cvt_f32_bf16_kernel<<<(wn4 + 255) / 256, 256, 0, stream>>>(Wv, Wvb, wn4);
  cvt_f32_bf16_kernel<<<(wn4 + 255) / 256, 256, 0, stream>>>(Wo, Wob, wn4);

  const dim3 gproj(BLROWS / 128, HH);  // (64, 16)
  const dim3 bproj(128);
  // Q pre-scale: softmax scale (1/8) * log2(e) so attention uses exp2.
  qkv_proj_rope_kernel<<<gproj, bproj, 0, stream>>>(Xqb, Wqb, bq, Qw, 1,
                                                    0.125f * LOG2E);
  qkv_proj_rope_kernel<<<gproj, bproj, 0, stream>>>(Xkb, Wkb, bk, Kw, 1, 1.0f);
  qkv_proj_rope_kernel<<<gproj, bproj, 0, stream>>>(Xvb, Wvb, bv, Vt, 0, 1.0f);

  attn_kernel<<<dim3(LL / 128, BB * HH), 256, 0, stream>>>(Qw, Kw, Vt, attnb);

  out_proj_kernel<<<gproj, bproj, 0, stream>>>(attnb, Wob, bo, (float*)d_out);
}
